// FusedStatefulQwen3Decoder_18554258718783
// MI455X (gfx1250) — matrix operand order repaired and checked by reading, not compiled, not run on hardware
//
#include <hip/hip_runtime.h>
#include <hip/hip_fp16.h>
#include <hip/hip_bf16.h>

// ---------------- model constants ----------------
#define Lyr   4
#define HID   2048
#define QH    16
#define KVH   8
#define HD    128
#define REP   (QH / KVH)      // 2
#define INTR  8192
#define VOCAB 32000
#define MAXS  512
#define QLEN  128
#define PAST  384
#define ENDS  512
#define SCALE 0.08838834764831845f  // 1/sqrt(128)
#define KSPLIT 8                    // LM-head K splits

typedef __attribute__((ext_vector_type(16))) _Float16 v16h_t;
typedef __attribute__((ext_vector_type(8)))  _Float16 v8h_t;
typedef __attribute__((ext_vector_type(8)))  float    v8f_t;

// ---------------- WMMA helpers (CDNA5 16x16x32 f16 -> f32) ----------------
__device__ __forceinline__ v8f_t wmma16(v16h_t a, v16h_t b, v8f_t c) {
  // (neg_a, A, neg_b, B, c_mod, C, reuse_a, reuse_b)
  return __builtin_amdgcn_wmma_f32_16x16x32_f16(false, a, false, b, (short)0, c,
                                                false, false);
}

// A fragment: 16(M) x 32(K), row-major source with leading dim lda (halves).
// lane 0-15: M=lane, halves 0..7 -> K 0..7, 8..15 -> K 16..23
// lane 16-31: M=lane-16, K shifted by +8 within each 16-group.
__device__ __forceinline__ v16h_t load_a_frag(const _Float16* A, int lda) {
  int lane = threadIdx.x & 31;
  int m  = lane & 15;
  int kb = (lane & 16) ? 8 : 0;
  const _Float16* base = A + (size_t)m * lda + kb;
  v8h_t lo = *(const v8h_t*)(base);        // K = kb..kb+7
  v8h_t hi = *(const v8h_t*)(base + 16);   // K = 16+kb..16+kb+7
  v16h_t r;
#pragma unroll
  for (int i = 0; i < 8; ++i) { r[i] = lo[i]; r[i + 8] = hi[i]; }
  return r;
}

// B fragment from an N-major (transposed) source: Src[n*ld + k] = B[k][n].
// lane 0-15: N=lane, halves 0..7 -> K 0..7, 8..15 -> K 16..23; lane 16-31: K shifted by +8.
// Two aligned 16B vector loads per lane.
__device__ __forceinline__ v16h_t load_bT_frag(const _Float16* Src, int ld) {
  int lane = threadIdx.x & 31;
  int n  = lane & 15;
  int kb = (lane & 16) ? 8 : 0;
  const _Float16* base = Src + (size_t)n * ld + kb;
  v8h_t lo = *(const v8h_t*)(base);
  v8h_t hi = *(const v8h_t*)(base + 16);
  v16h_t r;
#pragma unroll
  for (int i = 0; i < 8; ++i) { r[i] = lo[i]; r[i + 8] = hi[i]; }
  return r;
}

// ---------------- rmsnorm kernels ----------------
__global__ __launch_bounds__(256) void rmsnorm_f16_kernel(
    const float* __restrict__ x, const float* __restrict__ w,
    _Float16* __restrict__ out, int N) {
  int row = blockIdx.x;
  const float* xr = x + (size_t)row * N;
  __shared__ float red[8];
  float ss = 0.f;
  for (int i = threadIdx.x; i < N; i += 256) { float v = xr[i]; ss += v * v; }
#pragma unroll
  for (int off = 16; off > 0; off >>= 1) ss += __shfl_down(ss, off, 32);
  if ((threadIdx.x & 31) == 0) red[threadIdx.x >> 5] = ss;
  __syncthreads();
  if (threadIdx.x == 0) {
    float t = 0.f;
#pragma unroll
    for (int i = 0; i < 8; ++i) t += red[i];
    red[0] = rsqrtf(t / (float)N + 1e-6f);
  }
  __syncthreads();
  float rs = red[0];
  for (int i = threadIdx.x; i < N; i += 256)
    out[(size_t)row * N + i] = (_Float16)(xr[i] * rs * w[i]);
}

__global__ __launch_bounds__(256) void rmsnorm_f32_kernel(
    const float* __restrict__ x, const float* __restrict__ w,
    float* __restrict__ out, int N) {
  const float* xr = x;   // single row
  __shared__ float red[8];
  float ss = 0.f;
  for (int i = threadIdx.x; i < N; i += 256) { float v = xr[i]; ss += v * v; }
#pragma unroll
  for (int off = 16; off > 0; off >>= 1) ss += __shfl_down(ss, off, 32);
  if ((threadIdx.x & 31) == 0) red[threadIdx.x >> 5] = ss;
  __syncthreads();
  if (threadIdx.x == 0) {
    float t = 0.f;
#pragma unroll
    for (int i = 0; i < 8; ++i) t += red[i];
    red[0] = rsqrtf(t / (float)N + 1e-6f);
  }
  __syncthreads();
  float rs = red[0];
  for (int i = threadIdx.x; i < N; i += 256)
    out[i] = xr[i] * rs * w[i];
}

// ---------------- WMMA GEMM: C[128,N] = A_f16[128,K] * W_f32->f16[K,N] -----
// BM = 128 (full M): every weight tile fetched exactly once per block.
// Double-buffered LDS; per K-step each wave runs 4 WMMAs off one A fragment.
// All fragments loaded into distinct registers first so the ds_loads clause
// and the 4 WMMAs issue back-to-back (no per-WMMA LDS round-trip stall).
#define BM 128
#define BN 64
#define BK 32
#define BKP 48   // padded row (halves): 96B rows -> 16B-aligned b128, 2-way banks

__global__ __launch_bounds__(256) void gemm_wmma_kernel(
    const _Float16* __restrict__ A, const float* __restrict__ W,
    float* C, const float* residual, int N, int K) {
  __shared__ _Float16 As[2][BM][BK];     // 2 x 8 KB
  __shared__ _Float16 BsT[2][BN][BKP];   // 2 x 6 KB   (BsT[n][k] = B[k][n])
  int bn = blockIdx.x * BN;
  int tid  = threadIdx.x;
  int wave = tid >> 5;
  int wrow = wave * 16;           // wave's 16-row strip of M
  v8f_t acc0 = {}, acc1 = {}, acc2 = {}, acc3 = {};

  int arow = tid >> 1;            // 0..127
  int acol = (tid & 1) << 4;      // 0,16
  int brow = tid >> 3;            // 0..31  (k)
  int bcol = (tid & 7) << 3;      // 0..56  (n)

  // staging registers (global fetch decoupled from LDS store)
  v8h_t aR0, aR1;
  float4 w0, w1;

  auto global_fetch = [&](int k0) {
    const _Float16* ap = A + (size_t)arow * K + k0 + acol;
    aR0 = *(const v8h_t*)(ap);
    aR1 = *(const v8h_t*)(ap + 8);
    const float* wp = W + (size_t)(k0 + brow) * N + bn + bcol;
    w0 = *(const float4*)(wp);
    w1 = *(const float4*)(wp + 4);
  };
  auto lds_store = [&](int buf) {
    *(v8h_t*)&As[buf][arow][acol] = aR0;
    *(v8h_t*)&As[buf][arow][acol + 8] = aR1;
    BsT[buf][bcol + 0][brow] = (_Float16)w0.x;
    BsT[buf][bcol + 1][brow] = (_Float16)w0.y;
    BsT[buf][bcol + 2][brow] = (_Float16)w0.z;
    BsT[buf][bcol + 3][brow] = (_Float16)w0.w;
    BsT[buf][bcol + 4][brow] = (_Float16)w1.x;
    BsT[buf][bcol + 5][brow] = (_Float16)w1.y;
    BsT[buf][bcol + 6][brow] = (_Float16)w1.z;
    BsT[buf][bcol + 7][brow] = (_Float16)w1.w;
  };
  auto compute = [&](int buf) {
    // distinct registers for every fragment -> one dscnt wait, 4 WMMAs b2b
    v16h_t af  = load_a_frag(&As[buf][wrow][0], BK);
    v16h_t bf0 = load_bT_frag(&BsT[buf][0][0], BKP);
    v16h_t bf1 = load_bT_frag(&BsT[buf][16][0], BKP);
    v16h_t bf2 = load_bT_frag(&BsT[buf][32][0], BKP);
    v16h_t bf3 = load_bT_frag(&BsT[buf][48][0], BKP);
    acc0 = wmma16(af, bf0, acc0);
    acc1 = wmma16(af, bf1, acc1);
    acc2 = wmma16(af, bf2, acc2);
    acc3 = wmma16(af, bf3, acc3);
  };

  global_fetch(0);
  lds_store(0);
  __syncthreads();
  int buf = 0;
  for (int k0 = BK; k0 < K; k0 += BK) {
    global_fetch(k0);                 // next tile -> regs (overlaps WMMA)
    if (k0 + BK < K)                  // warm L2 for tile after next
      __builtin_prefetch(W + (size_t)(k0 + BK + brow) * N + bn + bcol, 0, 1);
    compute(buf);                     // WMMA from current buffer
    lds_store(buf ^ 1);               // commit next tile
    __syncthreads();
    buf ^= 1;
  }
  compute(buf);

  int lane = tid & 31;
  int nn = lane & 15;
  int mb = wrow + ((lane & 16) ? 8 : 0);
  v8f_t accs[4] = {acc0, acc1, acc2, acc3};
#pragma unroll
  for (int t = 0; t < 4; ++t) {
    int n0 = bn + t * 16 + nn;
#pragma unroll
    for (int r = 0; r < 8; ++r) {
      size_t i0 = (size_t)(mb + r) * N + n0;
      float v = accs[t][r];
      if (residual) v += residual[i0];
      C[i0] = v;
    }
  }
}

// ---------------- cache copy (old K/V -> contiguous f16 buffers) -----------
// K kept row-major [KVH][ENDS][HD]; V stored transposed [KVH][HD][ENDS] so
// the P*V WMMA B-fragments are contiguous vector loads.
__global__ __launch_bounds__(256) void cache_copy_kernel(
    const _Float16* __restrict__ kc, const _Float16* __restrict__ vc,
    _Float16* __restrict__ kbuf, _Float16* __restrict__ vbufT) {
  size_t idx = (size_t)blockIdx.x * 256 + threadIdx.x;
  size_t total = (size_t)KVH * PAST * HD;
  if (idx >= total) return;
  int d = idx % HD;
  size_t t = idx / HD;
  int pos = t % PAST;
  int h = t / PAST;
  size_t src = ((size_t)h * MAXS + pos) * HD + d;
  kbuf[((size_t)h * ENDS + pos) * HD + d] = kc[src];
  vbufT[((size_t)h * HD + d) * ENDS + pos] = vc[src];
}

// ---------------- q/k rmsnorm + RoPE + append to cache bufs ----------------
// blockIdx.x = token, blockIdx.y = head slot (0..QH-1:Q, QH..QH+KVH-1:K, rest:V)
__global__ __launch_bounds__(128) void rope_cache_kernel(
    const float* __restrict__ qproj, const float* __restrict__ kproj,
    const float* __restrict__ vproj, const float* __restrict__ qn,
    const float* __restrict__ kn, const float* __restrict__ cosb,
    const float* __restrict__ sinb, _Float16* __restrict__ qbuf,
    _Float16* __restrict__ kbuf, _Float16* __restrict__ vbufT) {
  int q = blockIdx.x;
  int hidx = blockIdx.y;
  int d = threadIdx.x;
  if (hidx >= QH + KVH) {  // V path: plain cast into transposed cache buffer
    int h = hidx - QH - KVH;
    vbufT[((size_t)h * HD + d) * ENDS + PAST + q] =
        (_Float16)vproj[((size_t)q * KVH + h) * HD + d];
    return;
  }
  __shared__ float red[4];
  __shared__ float vsh[HD];
  bool isQ = hidx < QH;
  const float* src = isQ ? (qproj + ((size_t)q * QH + hidx) * HD)
                         : (kproj + ((size_t)q * KVH + (hidx - QH)) * HD);
  float val = src[d];
  float ss = val * val;
#pragma unroll
  for (int off = 16; off > 0; off >>= 1) ss += __shfl_down(ss, off, 32);
  if ((threadIdx.x & 31) == 0) red[threadIdx.x >> 5] = ss;
  __syncthreads();
  float tot = red[0] + red[1] + red[2] + red[3];
  float rs = rsqrtf(tot / (float)HD + 1e-6f);
  const float* nw = isQ ? qn : kn;
  float x = val * rs * nw[d];
  vsh[d] = x;
  __syncthreads();
  float c = cosb[(size_t)q * HD + d];
  float s = sinb[(size_t)q * HD + d];
  float other = (d < HD / 2) ? -vsh[d + HD / 2] : vsh[d - HD / 2];
  float o = x * c + other * s;
  if (isQ)
    qbuf[((size_t)hidx * QLEN + q) * HD + d] = (_Float16)o;
  else
    kbuf[((size_t)(hidx - QH) * ENDS + PAST + q) * HD + d] = (_Float16)o;
}

// ---------------- attention: 1 wave = (head, 16-query tile) ----------------
__global__ __launch_bounds__(32) void attn_kernel(
    const _Float16* __restrict__ qbuf,   // [QH][QLEN][HD]
    const _Float16* __restrict__ kbuf,   // [KVH][ENDS][HD]
    const _Float16* __restrict__ vbufT,  // [KVH][HD][ENDS]
    const float* __restrict__ mask,      // [QLEN][ENDS]
    _Float16* __restrict__ obuf) {       // [QLEN][QH*HD]
  __shared__ float    S[16][ENDS];       // 32 KB
  __shared__ _Float16 P[16][ENDS];       // 16 KB
  int h  = blockIdx.x;
  int qt = blockIdx.y;
  int kvh = h / REP;
  const _Float16* Q  = qbuf + ((size_t)h * QLEN + qt * 16) * HD;
  const _Float16* Kc = kbuf + (size_t)kvh * ENDS * HD;
  const _Float16* Vt = vbufT + (size_t)kvh * HD * ENDS;
  int lane = threadIdx.x & 31;
  int nn = lane & 15;
  int mb = (lane & 16) ? 8 : 0;

  // --- scores: S = (Q K^T)*scale + mask ---
  v16h_t aq[4];
#pragma unroll
  for (int c = 0; c < 4; ++c) aq[c] = load_a_frag(Q + c * 32, HD);
  for (int n0 = 0; n0 < ENDS; n0 += 16) {
    // B[k=d][n=pos] = Kc[pos*HD + d]; distinct regs -> loads clause together
    v16h_t b0 = load_bT_frag(Kc + (size_t)n0 * HD + 0, HD);
    v16h_t b1 = load_bT_frag(Kc + (size_t)n0 * HD + 32, HD);
    v16h_t b2 = load_bT_frag(Kc + (size_t)n0 * HD + 64, HD);
    v16h_t b3 = load_bT_frag(Kc + (size_t)n0 * HD + 96, HD);
    v8f_t acc = {};
    acc = wmma16(aq[0], b0, acc);
    acc = wmma16(aq[1], b1, acc);
    acc = wmma16(aq[2], b2, acc);
    acc = wmma16(aq[3], b3, acc);
#pragma unroll
    for (int r = 0; r < 8; ++r) {
      int m = mb + r;
      S[m][n0 + nn] =
          acc[r] * SCALE + mask[(size_t)(qt * 16 + m) * ENDS + n0 + nn];
    }
  }
  __syncthreads();

  // --- softmax over 512 cols: lane pair (l, l+16) splits a row ---
  {
    int row = lane & 15;
    int cb = (lane & 16) ? (ENDS / 2) : 0;
    float mx = -1e30f;
    for (int c = 0; c < ENDS / 2; ++c) mx = fmaxf(mx, S[row][cb + c]);
    mx = fmaxf(mx, __shfl_xor(mx, 16, 32));
    float sum = 0.f;
    for (int c = 0; c < ENDS / 2; ++c) {
      float e = __expf(S[row][cb + c] - mx);
      S[row][cb + c] = e;
      sum += e;
    }
    sum += __shfl_xor(sum, 16, 32);
    float inv = 1.0f / sum;
    for (int c = 0; c < ENDS / 2; ++c)
      P[row][cb + c] = (_Float16)(S[row][cb + c] * inv);
  }
  __syncthreads();

  // --- O = P V  (2x unrolled, distinct regs per step) ---
#pragma unroll
  for (int nt = 0; nt < HD / 16; ++nt) {
    v8f_t acc = {};
    for (int kc = 0; kc < ENDS; kc += 64) {
      v16h_t a0 = load_a_frag(&P[0][0] + kc, ENDS);
      v16h_t b0 = load_bT_frag(Vt + (size_t)(nt * 16) * ENDS + kc, ENDS);
      v16h_t a1 = load_a_frag(&P[0][0] + kc + 32, ENDS);
      v16h_t b1 = load_bT_frag(Vt + (size_t)(nt * 16) * ENDS + kc + 32, ENDS);
      acc = wmma16(a0, b0, acc);
      acc = wmma16(a1, b1, acc);
    }
#pragma unroll
    for (int r = 0; r < 8; ++r) {
      int m = mb + r;
      obuf[(size_t)(qt * 16 + m) * (QH * HD) + (size_t)h * HD + nt * 16 + nn] =
          (_Float16)acc[r];
    }
  }
}

// ---------------- SiLU(g) * u -> f16 ----------------
__global__ __launch_bounds__(256) void silu_mul_kernel(
    const float* __restrict__ g, const float* __restrict__ u,
    _Float16* __restrict__ t, size_t n) {
  size_t i = (size_t)blockIdx.x * 256 + threadIdx.x;
  if (i >= n) return;
  float gv = g[i];
  float sv = gv / (1.0f + __expf(-gv));
  t[i] = (_Float16)(sv * u[i]);
}

// ---------------- LM head: K-split GEMV + deterministic reduce -------------
__global__ __launch_bounds__(256) void lmhead_partial_kernel(
    const float* __restrict__ x, const float* __restrict__ Wlm,
    float* __restrict__ partial) {  // [KSPLIT][VOCAB]
  int n = blockIdx.x * 256 + threadIdx.x;
  int ks = blockIdx.y;
  if (n >= VOCAB) return;
  int kb = ks * (HID / KSPLIT);
  float acc = 0.f;
#pragma unroll 8
  for (int k = 0; k < HID / KSPLIT; ++k)
    acc += x[kb + k] * Wlm[(size_t)(kb + k) * VOCAB + n];
  partial[(size_t)ks * VOCAB + n] = acc;
}

__global__ __launch_bounds__(256) void lmhead_reduce_kernel(
    const float* __restrict__ partial, float* __restrict__ out) {
  int n = blockIdx.x * 256 + threadIdx.x;
  if (n >= VOCAB) return;
  float acc = 0.f;
#pragma unroll
  for (int ks = 0; ks < KSPLIT; ++ks) acc += partial[(size_t)ks * VOCAB + n];
  out[n] = acc;
}

// ---------------- host orchestration ----------------
extern "C" void kernel_launch(void* const* d_in, const int* in_sizes, int n_in,
                              void* d_out, int out_size, void* d_ws,
                              size_t ws_size, hipStream_t stream) {
  const float*    hidden = (const float*)d_in[0];
  const float*    cosb   = (const float*)d_in[1];
  const float*    sinb   = (const float*)d_in[2];
  const float*    mask   = (const float*)d_in[3];
  const _Float16* kcache = (const _Float16*)d_in[4];
  const _Float16* vcache = (const _Float16*)d_in[5];
  const float* ln1 = (const float*)d_in[6];
  const float* qn  = (const float*)d_in[7];
  const float* kn  = (const float*)d_in[8];
  const float* Wq  = (const float*)d_in[9];
  const float* Wk  = (const float*)d_in[10];
  const float* Wv  = (const float*)d_in[11];
  const float* Wo  = (const float*)d_in[12];
  const float* ln2 = (const float*)d_in[13];
  const float* Wg  = (const float*)d_in[14];
  const float* Wu  = (const float*)d_in[15];
  const float* Wd  = (const float*)d_in[16];
  const float* fnw = (const float*)d_in[17];
  const float* Wlm = (const float*)d_in[18];
  float* out = (float*)d_out;

  char* p = (char*)d_ws;
  auto alloc = [&](size_t bytes) -> void* {
    void* r = (void*)p;
    p += (bytes + 255) & ~(size_t)255;
    return r;
  };
  float*    h     = (float*)alloc((size_t)QLEN * HID * 4);
  _Float16* xn    = (_Float16*)alloc((size_t)QLEN * HID * 2);
  float*    qproj = (float*)alloc((size_t)QLEN * QH * HD * 4);
  float*    kproj = (float*)alloc((size_t)QLEN * KVH * HD * 4);
  float*    vproj = (float*)alloc((size_t)QLEN * KVH * HD * 4);
  _Float16* qbuf  = (_Float16*)alloc((size_t)QH * QLEN * HD * 2);
  _Float16* kbuf  = (_Float16*)alloc((size_t)KVH * ENDS * HD * 2);
  _Float16* vbufT = (_Float16*)alloc((size_t)KVH * HD * ENDS * 2);
  _Float16* obuf  = (_Float16*)alloc((size_t)QLEN * QH * HD * 2);
  float*    gbuf  = (float*)alloc((size_t)QLEN * INTR * 4);
  float*    ubuf  = (float*)alloc((size_t)QLEN * INTR * 4);
  _Float16* tbuf  = (_Float16*)alloc((size_t)QLEN * INTR * 2);
  float*    xlast = (float*)alloc((size_t)HID * 4);
  float*    lmp   = (float*)alloc((size_t)KSPLIT * VOCAB * 4);

  hipMemcpyAsync(h, hidden, (size_t)QLEN * HID * sizeof(float),
                 hipMemcpyDeviceToDevice, stream);

  for (int l = 0; l < Lyr; ++l) {
    const float* ln1w = ln1 + (size_t)l * HID;
    const float* qnw  = qn + (size_t)l * HD;
    const float* knw  = kn + (size_t)l * HD;
    const float* wq   = Wq + (size_t)l * HID * QH * HD;
    const float* wk   = Wk + (size_t)l * HID * KVH * HD;
    const float* wv   = Wv + (size_t)l * HID * KVH * HD;
    const float* wo   = Wo + (size_t)l * QH * HD * HID;
    const float* ln2w = ln2 + (size_t)l * HID;
    const float* wg   = Wg + (size_t)l * HID * INTR;
    const float* wu   = Wu + (size_t)l * HID * INTR;
    const float* wd   = Wd + (size_t)l * INTR * HID;
    const _Float16* kc = kcache + (size_t)l * KVH * MAXS * HD;
    const _Float16* vc = vcache + (size_t)l * KVH * MAXS * HD;

    // x = rmsnorm(h, ln1) -> f16
    rmsnorm_f16_kernel<<<QLEN, 256, 0, stream>>>(h, ln1w, xn, HID);
    // q/k/v projections (WMMA, fp32 weights converted on the fly)
    gemm_wmma_kernel<<<(QH * HD) / BN, 256, 0, stream>>>(
        xn, wq, qproj, nullptr, QH * HD, HID);
    gemm_wmma_kernel<<<(KVH * HD) / BN, 256, 0, stream>>>(
        xn, wk, kproj, nullptr, KVH * HD, HID);
    gemm_wmma_kernel<<<(KVH * HD) / BN, 256, 0, stream>>>(
        xn, wv, vproj, nullptr, KVH * HD, HID);
    // assemble f16 K/V buffers (K row-major, V transposed)
    {
      size_t total = (size_t)KVH * PAST * HD;
      cache_copy_kernel<<<(unsigned)((total + 255) / 256), 256, 0, stream>>>(
          kc, vc, kbuf, vbufT);
    }
    rope_cache_kernel<<<dim3(QLEN, QH + 2 * KVH), 128, 0, stream>>>(
        qproj, kproj, vproj, qnw, knw, cosb, sinb, qbuf, kbuf, vbufT);
    // attention (WMMA QK^T + softmax + WMMA PV)
    attn_kernel<<<dim3(QH, QLEN / 16), 32, 0, stream>>>(qbuf, kbuf, vbufT,
                                                        mask, obuf);
    // h += attn @ Wo  (fused residual)
    gemm_wmma_kernel<<<HID / BN, 256, 0, stream>>>(
        obuf, wo, h, h, HID, QH * HD);
    // MLP
    rmsnorm_f16_kernel<<<QLEN, 256, 0, stream>>>(h, ln2w, xn, HID);
    gemm_wmma_kernel<<<INTR / BN, 256, 0, stream>>>(
        xn, wg, gbuf, nullptr, INTR, HID);
    gemm_wmma_kernel<<<INTR / BN, 256, 0, stream>>>(
        xn, wu, ubuf, nullptr, INTR, HID);
    {
      size_t total = (size_t)QLEN * INTR;
      silu_mul_kernel<<<(unsigned)((total + 255) / 256), 256, 0, stream>>>(
          gbuf, ubuf, tbuf, total);
    }
    // h += t @ Wd  (fused residual)
    gemm_wmma_kernel<<<HID / BN, 256, 0, stream>>>(
        tbuf, wd, h, h, HID, INTR);
  }

  // final norm of last token + K-split LM head (deterministic reduce)
  rmsnorm_f32_kernel<<<1, 256, 0, stream>>>(h + (size_t)(QLEN - 1) * HID, fnw,
                                            xlast, HID);
  lmhead_partial_kernel<<<dim3((VOCAB + 255) / 256, KSPLIT), 256, 0, stream>>>(
      xlast, Wlm, lmp);
  lmhead_reduce_kernel<<<(VOCAB + 255) / 256, 256, 0, stream>>>(lmp, out);
}
